// DualEncoderWithVQ_17188459118924
// MI455X (gfx1250) — matrix operand order repaired and checked by reading, not compiled, not run on hardware
//
#include <hip/hip_runtime.h>
#include <cmath>

// ---------------------------------------------------------------------------
// CDNA5 (gfx1250) VQ-VAE forward. All GEMM-like work (convs, convT, einsum)
// runs on v_wmma_f32_16x16x32_bf16 (bf16 operands, f32 accumulate).
// 128-thread blocks (4 waves) compute a 32x32 macro-tile; tiles are staged in
// LDS in *fragment order* so each operand is a single 32B (2x ds_load_b128)
// load per wave; global loads for the next k-step are software-pipelined.
// In-loop index decompose uses host-precomputed magic-number division; 1x1
// convs take a decompose-free linear-stream path; input activations can be
// applied on load (fused LeakyReLU) so no standalone elementwise passes.
// ---------------------------------------------------------------------------

typedef __attribute__((ext_vector_type(16))) __bf16 v16bf;
typedef __attribute__((ext_vector_type(8)))  float v8f;

__device__ __forceinline__ float apply_act(float v, int act) {
  if (act == 1) return v >= 0.f ? v : 0.01f * v;  // LeakyReLU(0.01)
  if (act == 2) return v > 0.f ? v : 0.f;          // ReLU
  if (act == 3) return tanhf(v);
  return v;
}

// Exact unsigned division by small runtime constant via magic multiply.
// m = ceil(2^32 / d) (host-side); exact for the k ranges used here (k < 2^16).
__device__ __forceinline__ int magic_div(int x, unsigned m) {
  return (int)__umulhi((unsigned)x, m);
}

// Fragment-order helpers (ISA 16-bit A 16x32 / B 32x16 VGPR layouts):
//   A slot (dl, e): m = dl&15,  k = (e<8?e:e+8) + (dl>>4)*8
//   B slot (dl, e): n = dl&15,  k = (e<8?e:e+8) + (dl>>4)*8
__device__ __forceinline__ int fragA_k(int dl, int e) {
  return (e < 8 ? e : e + 8) + (dl >> 4) * 8;
}
__device__ __forceinline__ int fragB_k(int dl, int e) {
  return (e < 8 ? e : e + 8) + (dl >> 4) * 8;
}

// ---------------------------------------------------------------------------
// Generic implicit-GEMM conv / transposed-conv.
//   M = Cout, Ncols = N*Hout*Wout, K = Cin*KH*KW
//   mode 0: conv, weights OIHW (contiguous along K), iy = oy*stride - pad + ky
//   mode 1: convT stride 2, weights [Cin][Cout][KH][KW], iy = (oy+pad-ky)/2
// Block: 128 threads = 4 waves; macro tile 32(M) x 32(N); wave (msub,nsub).
// ---------------------------------------------------------------------------
struct ConvP {
  const float* X; const float* W; const float* B; const float* R; float* Y;
  int N, Cin, Hin, Win, Cout, Hout, Wout, KH, KW, stride, pad, mode, act;
  int bact;            // activation applied to input values on load
  int simple;          // 1x1 / s1 / p0 / mode0: decompose-free gather
  unsigned mKhkw;      // ceil(2^32 / (KH*KW)), valid when KH*KW > 1
  unsigned mKW;        // ceil(2^32 / KW),      valid when KW > 1
};

__global__ __launch_bounds__(128) void conv_wmma_kernel(ConvP p) {
  __shared__ __align__(32) __bf16 As[1024];   // 2 M-subtiles, fragment order
  __shared__ __align__(32) __bf16 Bs[1024];   // 2 N-subtiles, fragment order

  const int t     = threadIdx.x;
  const int lane  = t & 31;
  const int wave  = t >> 5;
  const int msub  = wave >> 1;
  const int nsub  = wave & 1;
  const int tm    = blockIdx.x;
  const int tn    = blockIdx.y;
  const int M     = p.Cout;
  const int HWo   = p.Hout * p.Wout;
  const int HWin  = p.Hin * p.Win;
  const int Ncols = p.N * HWo;
  const int khkw  = p.KH * p.KW;
  const int K     = p.Cin * khkw;

  // --- per-thread staging slot maps (k-invariant) -------------------------
  int asl[8], akk[8], aco[8];
  int bsl[8], bkk[8], bb[8], boy[8], box[8], bvalid[8];
  long bsimple[8];
#pragma unroll
  for (int j = 0; j < 8; ++j) {
    int s = t + 128 * j;               // 0..1023
    int sub = s >> 9, r = s & 511, dl = r & 31, e = r >> 5;
    int slot = sub * 512 + dl * 16 + e;
    asl[j] = slot;
    akk[j] = fragA_k(dl, e);
    aco[j] = tm * 32 + (dl & 15) + sub * 16;
    bsl[j] = slot;
    bkk[j] = fragB_k(dl, e);
    int col = tn * 32 + (dl & 15) + sub * 16;
    bvalid[j] = (col < Ncols);
    int cc = bvalid[j] ? col : 0;
    bb[j] = cc / HWo;
    int r2 = cc % HWo;
    boy[j] = r2 / p.Wout;
    box[j] = r2 % p.Wout;
    bsimple[j] = (long)bb[j] * p.Cin * HWin + (long)boy[j] * p.Win + box[j];
  }

  auto loadA = [&](int k0, float* ar) {
#pragma unroll
    for (int j = 0; j < 8; ++j) {
      int k = k0 + akk[j];
      int co = aco[j];
      float v = 0.f;
      if (k < K && co < M) {
        if (p.mode == 0) {
          v = p.W[(long)co * K + k];               // OIHW: contiguous along K
        } else {
          int ci = magic_div(k, p.mKhkw);
          int r  = k - ci * khkw;
          int ky = magic_div(r, p.mKW);
          int kx = r - ky * p.KW;
          v = p.W[(((long)ci * p.Cout + co) * p.KH + ky) * p.KW + kx];
        }
      }
      ar[j] = v;
    }
  };
  auto loadB = [&](int k0, float* br) {
#pragma unroll
    for (int j = 0; j < 8; ++j) {
      int k = k0 + bkk[j];
      float v = 0.f;
      if (k < K && bvalid[j]) {
        if (p.simple) {
          v = p.X[bsimple[j] + (long)k * HWin];    // 1x1/s1/p0: linear stream
        } else {
          int ci = magic_div(k, p.mKhkw);
          int r  = k - ci * khkw;
          int ky = magic_div(r, p.mKW);
          int kx = r - ky * p.KW;
          int iy, ix; bool ok;
          if (p.mode == 0) {
            iy = boy[j] * p.stride - p.pad + ky;
            ix = box[j] * p.stride - p.pad + kx;
            ok = (iy >= 0) && (iy < p.Hin) && (ix >= 0) && (ix < p.Win);
          } else {
            int ty = boy[j] + p.pad - ky, tx = box[j] + p.pad - kx;
            ok = (ty >= 0) && (tx >= 0) && !(ty & 1) && !(tx & 1);
            iy = ty >> 1; ix = tx >> 1;
            ok = ok && (iy < p.Hin) && (ix < p.Win);
          }
          if (ok) v = p.X[(((long)bb[j] * p.Cin + ci) * p.Hin + iy) * p.Win + ix];
        }
        v = apply_act(v, p.bact);                  // fused input activation
      }
      br[j] = v;
    }
  };

  // --- software-pipelined K loop ------------------------------------------
  float arb[8], brb[8];
  loadA(0, arb);
  loadB(0, brb);
  v8f acc = {};
  for (int k0 = 0; k0 < K; k0 += 32) {
#pragma unroll
    for (int j = 0; j < 8; ++j) As[asl[j]] = (__bf16)arb[j];
#pragma unroll
    for (int j = 0; j < 8; ++j) Bs[bsl[j]] = (__bf16)brb[j];
    __syncthreads();
    if (k0 + 32 < K) { loadA(k0 + 32, arb); loadB(k0 + 32, brb); }
    v16bf a = *(const v16bf*)(As + msub * 512 + lane * 16);
    v16bf b = *(const v16bf*)(Bs + nsub * 512 + lane * 16);
    acc = __builtin_amdgcn_wmma_f32_16x16x32_bf16(false, a, false, b,
                                                  (short)0, acc, false, false);
    __syncthreads();
  }

  // --- epilogue: C/D layout lane n = lane&15, row m = r + 8*(lane>>4) -----
  const int n   = lane & 15;
  const int hi  = lane >> 4;
  const int col = tn * 32 + nsub * 16 + n;
  if (col < Ncols) {
    int b = col / HWo, r2 = col % HWo, oy = r2 / p.Wout, ox = r2 % p.Wout;
#pragma unroll
    for (int r = 0; r < 8; ++r) {
      int co = tm * 32 + msub * 16 + r + hi * 8;
      if (co < M) {
        float v = acc[r];
        if (p.B) v += p.B[co];
        long oidx = (((long)b * p.Cout + co) * p.Hout + oy) * p.Wout + ox;
        if (p.R) v += p.R[oidx];
        p.Y[oidx] = apply_act(v, p.act);
      }
    }
  }
}

// ---------------------------------------------------------------------------
// Per-token einsum: ez[c] = zc[c] @ emb_z_w[c] + b ; c in 0..255
//   A[16x64]: A[b][d] = Z[b][d][c]   (Z = dec_in [B,128,256], channels 0..63)
//   B[64x4096]: emb_z_w[c].  Macro tile 16(M) x 64(N): 4 waves share A.
// Output written directly into xc concat buffer at channel 3+c.
// ---------------------------------------------------------------------------
__global__ __launch_bounds__(128) void einsum_wmma_kernel(const float* __restrict__ Z,
                                                          const float* __restrict__ Wz,
                                                          const float* __restrict__ Bz,
                                                          float* __restrict__ XC) {
  __shared__ __align__(32) __bf16 As[512];        // one 16x32 A tile
  __shared__ __align__(32) __bf16 Bs[4 * 512];    // four 32x16 B subtiles

  const int t    = threadIdx.x;
  const int lane = t & 31;
  const int nsub = t >> 5;          // wave id = N subtile 0..3
  const int c    = blockIdx.x;      // token 0..255
  const int tn   = blockIdx.y;      // 0..63 (4096 / 64)

  v8f acc = {};
  for (int k0 = 0; k0 < 64; k0 += 32) {
#pragma unroll
    for (int j = 0; j < 4; ++j) {
      int s = t + 128 * j, dl = s & 31, e = s >> 5;
      int d = k0 + fragA_k(dl, e);
      int brow = dl & 15;
      As[dl * 16 + e] = (__bf16)Z[((long)brow * 128 + d) * 256 + c];
    }
#pragma unroll
    for (int j = 0; j < 16; ++j) {
      int s = t + 128 * j;
      int sub = s >> 9, r = s & 511, dl = r & 31, e = r >> 5;
      int d = k0 + fragB_k(dl, e);
      int col = tn * 64 + sub * 16 + (dl & 15);
      Bs[sub * 512 + dl * 16 + e] = (__bf16)Wz[((long)c * 64 + d) * 4096 + col];
    }
    __syncthreads();
    v16bf a = *(const v16bf*)(As + lane * 16);
    v16bf b = *(const v16bf*)(Bs + nsub * 512 + lane * 16);
    acc = __builtin_amdgcn_wmma_f32_16x16x32_bf16(false, a, false, b,
                                                  (short)0, acc, false, false);
    __syncthreads();
  }
  const int n   = lane & 15;
  const int hi  = lane >> 4;
  const int col = tn * 64 + nsub * 16 + n;
  const float bias = Bz[(long)c * 4096 + col];
#pragma unroll
  for (int r = 0; r < 8; ++r) {
    int b = r + hi * 8;                            // batch 0..15
    XC[(((long)b * 259 + 3 + c) * 4096) + col] = acc[r] + bias;
  }
}

// ---------------------------------------------------------------------------
// emb_x 1x1 conv (3->3) into xc channels 0..2
// ---------------------------------------------------------------------------
__global__ __launch_bounds__(256) void emb_x_kernel(const float* __restrict__ X,
                                                    const float* __restrict__ W,
                                                    const float* __restrict__ Bb,
                                                    float* __restrict__ XC) {
  int idx = blockIdx.x * blockDim.x + threadIdx.x;   // b*4096 + hw
  if (idx >= 16 * 4096) return;
  int b = idx >> 12, hw = idx & 4095;
  float x0 = X[((long)b * 3 + 0) * 4096 + hw];
  float x1 = X[((long)b * 3 + 1) * 4096 + hw];
  float x2 = X[((long)b * 3 + 2) * 4096 + hw];
#pragma unroll
  for (int co = 0; co < 3; ++co) {
    float v = W[co * 3 + 0] * x0 + W[co * 3 + 1] * x1 + W[co * 3 + 2] * x2 + Bb[co];
    XC[((long)b * 259 + co) * 4096 + hw] = v;
  }
}

// ---------------------------------------------------------------------------
// Vector quantization: nearest of 512 codes (D=64), gather into concat buffer,
// loss += (1+beta)*mean((q-lat)^2). Straight-through value == q.
// ---------------------------------------------------------------------------
#define VQ_LOSS_SCALE (1.25f / 262144.0f)   // (1+0.25) / (B*H*W*D)

__global__ __launch_bounds__(256) void vq_kernel(const float* __restrict__ Lat,
                                                 const float* __restrict__ E,
                                                 float* __restrict__ Dst,
                                                 int chOff,
                                                 float* __restrict__ lossAcc) {
  __shared__ float Ec[32 * 64];              // 32-code chunk of codebook
  const int token = blockIdx.x * 256 + threadIdx.x;   // 4096 tokens
  const int b = token >> 8, hw = token & 255;

  float lv[64];
#pragma unroll
  for (int d = 0; d < 64; ++d) lv[d] = Lat[((long)b * 64 + d) * 256 + hw];

  float best = 3.0e38f;
  int bi = 0;
  for (int c0 = 0; c0 < 512; c0 += 32) {
    __syncthreads();
    for (int i = threadIdx.x; i < 32 * 64; i += 256) Ec[i] = E[(long)c0 * 64 + i];
    __syncthreads();
    for (int c = 0; c < 32; ++c) {
      float d2 = 0.f;
#pragma unroll
      for (int d = 0; d < 64; ++d) {
        float tdiff = lv[d] - Ec[c * 64 + d];
        d2 += tdiff * tdiff;
      }
      if (d2 < best) { best = d2; bi = c0 + c; }
    }
  }
  float err = 0.f;
#pragma unroll
  for (int d = 0; d < 64; ++d) {
    float e = E[(long)bi * 64 + d];
    float tdiff = e - lv[d];
    err += tdiff * tdiff;
    Dst[((long)b * 128 + chOff + d) * 256 + hw] = e;   // straight-through == q
  }
  atomicAdd(lossAcc, err * VQ_LOSS_SCALE);
}

__global__ void zero_kernel(float* p) { if (threadIdx.x == 0) p[0] = 0.f; }

// ---------------------------------------------------------------------------
// Host side
// ---------------------------------------------------------------------------
extern "C" void kernel_launch(void* const* d_in, const int* in_sizes, int n_in,
                              void* d_out, int out_size, void* d_ws, size_t ws_size,
                              hipStream_t stream) {
  (void)n_in; (void)out_size; (void)ws_size;
  const float* x = (const float*)d_in[0];

  // Nested-dict flattening convention: jax-pytree sorts dict keys ('conv' first,
  // 589824 elems); plain insertion order puts down0_w first (6144 elems).
  const bool sortedKeys = (in_sizes[1] == 256 * 256 * 3 * 3);

  struct Enc { const float *d0w,*d0b,*d1w,*d1b,*cw,*cb,*ow,*ob,*ra[6],*rb[6]; };
  auto getEnc = [&](int base) {
    Enc e; auto P = [&](int o) { return (const float*)d_in[base + o]; };
    if (sortedKeys) { // keys: conv, down, out, res
      e.cw = P(0); e.cb = P(1); e.d0w = P(2); e.d0b = P(3);
      e.d1w = P(4); e.d1b = P(5); e.ow = P(6); e.ob = P(7);
      for (int i = 0; i < 6; ++i) { e.ra[i] = P(8 + 2*i); e.rb[i] = P(9 + 2*i); }
    } else {          // keys: down, res, conv, out
      e.d0w = P(0); e.d0b = P(1); e.d1w = P(2); e.d1b = P(3);
      for (int i = 0; i < 6; ++i) { e.ra[i] = P(4 + 2*i); e.rb[i] = P(5 + 2*i); }
      e.cw = P(16); e.cb = P(17); e.ow = P(18); e.ob = P(19);
    }
    return e;
  };
  Enc ez = getEnc(1), ec = getEnc(26);
  const float* cbz = (const float*)d_in[21];
  const float* ezw = (const float*)d_in[22];
  const float* ezb = (const float*)d_in[23];
  const float* exw = (const float*)d_in[24];
  const float* exb = (const float*)d_in[25];
  const float* cbc = (const float*)d_in[46];

  struct Dec { const float *iw,*ib,*ow,*ob,*ra[6],*rb[6],*uw,*ub; } dc;
  {
    int base = 47; auto P = [&](int o) { return (const float*)d_in[base + o]; };
    if (sortedKeys) { // keys: in, out, res, up
      dc.iw = P(0); dc.ib = P(1); dc.ow = P(2); dc.ob = P(3);
      for (int i = 0; i < 6; ++i) { dc.ra[i] = P(4 + 2*i); dc.rb[i] = P(5 + 2*i); }
      dc.uw = P(16); dc.ub = P(17);
    } else {          // keys: res, up, in, out
      for (int i = 0; i < 6; ++i) { dc.ra[i] = P(0 + 2*i); dc.rb[i] = P(1 + 2*i); }
      dc.uw = P(12); dc.ub = P(13); dc.iw = P(14); dc.ib = P(15);
      dc.ow = P(16); dc.ob = P(17);
    }
  }

  // Workspace layout (floats)
  float* ws    = (float*)d_ws;
  float* A1    = ws;                      // [16,128,32,32]  2,097,152
  float* A2    = A1    + 2097152;         // [16,256,16,16]  1,048,576
  float* Ff    = A2    + 1048576;         // [16,256,16,16]
  float* Tt    = Ff    + 1048576;         // [16,256,16,16]
  float* LAT   = Tt    + 1048576;         // [16, 64,16,16]    262,144
  float* DECIN = LAT   + 262144;          // [16,128,16,16]    524,288
  float* XC    = DECIN + 524288;          // [16,259,64,64] 16,973,824
  float* xhat  = (float*)d_out;
  float* loss  = xhat + 16 * 3 * 64 * 64;

  auto magic = [](int d) -> unsigned {
    return (d > 1) ? (unsigned)((0x100000000ULL + (unsigned)d - 1) / (unsigned)d) : 0u;
  };

  auto conv = [&](const float* X, const float* W, const float* B, const float* R,
                  float* Y, int N, int Cin, int Hin, int Win, int Cout,
                  int Hout, int Wout, int KH, int KW, int st, int pad,
                  int mode, int act, int bact) {
    int simple = (mode == 0 && KH == 1 && KW == 1 && st == 1 && pad == 0) ? 1 : 0;
    ConvP p{X, W, B, R, Y, N, Cin, Hin, Win, Cout, Hout, Wout, KH, KW, st, pad,
            mode, act, bact, simple, magic(KH * KW), magic(KW)};
    dim3 g((unsigned)((Cout + 31) / 32), (unsigned)((N * Hout * Wout + 31) / 32));
    hipLaunchKernelGGL(conv_wmma_kernel, g, dim3(128), 0, stream, p);
  };

  auto runEnc = [&](const float* in, int Cin, const Enc& e, float* lat) {
    conv(in, e.d0w, e.d0b, nullptr, A1, 16, Cin, 64, 64, 128, 32, 32, 4, 4, 2, 1, 0, 1, 0);
    conv(A1, e.d1w, e.d1b, nullptr, A2, 16, 128, 32, 32, 256, 16, 16, 4, 4, 2, 1, 0, 1, 0);
    conv(A2, e.cw,  e.cb,  nullptr, Ff, 16, 256, 16, 16, 256, 16, 16, 3, 3, 1, 1, 0, 1, 0);
    for (int i = 0; i < 6; ++i) {
      conv(Ff, e.ra[i], nullptr, nullptr, Tt, 16, 256, 16, 16, 256, 16, 16, 3, 3, 1, 1, 0, 2, 0);
      conv(Tt, e.rb[i], nullptr, Ff,      Ff, 16, 256, 16, 16, 256, 16, 16, 1, 1, 1, 0, 0, 0, 0);
    }
    // final lrelu fused into B-gather of the out conv (bact = 1)
    conv(Ff, e.ow, e.ob, nullptr, lat, 16, 256, 16, 16, 64, 16, 16, 1, 1, 1, 0, 0, 1, 1);
  };

  hipLaunchKernelGGL(zero_kernel, dim3(1), dim3(1), 0, stream, loss);

  // ---- branch Z ----
  runEnc(x, 3, ez, LAT);
  hipLaunchKernelGGL(vq_kernel, dim3(16), dim3(256), 0, stream, LAT, cbz, DECIN, 0, loss);

  // ---- conditioning input xc = concat(emb_x(x), einsum(z_q)) ----
  hipLaunchKernelGGL(einsum_wmma_kernel, dim3(256, 64), dim3(128), 0, stream,
                     DECIN, ezw, ezb, XC);
  hipLaunchKernelGGL(emb_x_kernel, dim3(256), dim3(256), 0, stream, x, exw, exb, XC);

  // ---- branch C ----
  runEnc(XC, 259, ec, LAT);
  hipLaunchKernelGGL(vq_kernel, dim3(16), dim3(256), 0, stream, LAT, cbc, DECIN, 64, loss);

  // ---- decoder on concat(z_q, c_q) [16,128,16,16] ----
  conv(DECIN, dc.iw, dc.ib, nullptr, Ff, 16, 128, 16, 16, 256, 16, 16, 3, 3, 1, 1, 0, 1, 0);
  for (int i = 0; i < 6; ++i) {
    conv(Ff, dc.ra[i], nullptr, nullptr, Tt, 16, 256, 16, 16, 256, 16, 16, 3, 3, 1, 1, 0, 2, 0);
    conv(Tt, dc.rb[i], nullptr, Ff,      Ff, 16, 256, 16, 16, 256, 16, 16, 1, 1, 1, 0, 0, 0, 0);
  }
  // final decoder lrelu fused into the up-convT's input gather (bact = 1)
  conv(Ff, dc.uw, dc.ub, nullptr, A1,   16, 256, 16, 16, 128, 32, 32, 4, 4, 2, 1, 1, 1, 1);
  conv(A1, dc.ow, dc.ob, nullptr, xhat, 16, 128, 32, 32,   3, 64, 64, 4, 4, 2, 1, 1, 3, 0);
}
